// MLPConv4d_23502061043775
// MI455X (gfx1250) — compile-verified, matrix operand order checked by reading, not run on hardware
//
#include <hip/hip_runtime.h>

// ---------------------------------------------------------------------------
// MLPConv4d on gfx1250: conv4d(s=2,p=1) -> GELU -> conv4d(s=2,p=1)
// Implicit-im2col GEMM with v_wmma_f32_16x16x32_bf16 (fp32 accumulate),
// double-buffered LDS slabs filled with global_load_async_to_lds_b128
// (ASYNCcnt) so the global->LDS copy hides under WMMA compute.
// ---------------------------------------------------------------------------

typedef __attribute__((ext_vector_type(16))) __bf16        v16bf;
typedef __attribute__((ext_vector_type(8)))  float         v8f;
typedef __attribute__((ext_vector_type(4)))  unsigned int  v4u;

union BFrag { v16bf v; v4u u[2]; };

__device__ __forceinline__ unsigned short f2bf(float f) {
  unsigned u = __float_as_uint(f);
  u += 0x7FFFu + ((u >> 16) & 1u);          // round-to-nearest-even
  return (unsigned short)(u >> 16);
}

// Async DMA, GV form: per-lane 64-bit global address -> LDS (b128 per lane).
__device__ __forceinline__ void async_b128(unsigned lds_off, const void* gptr) {
  const unsigned long long ga = (unsigned long long)gptr;
  asm volatile("global_load_async_to_lds_b128 %0, %1, off"
               :: "v"(lds_off), "v"(ga) : "memory");
}
__device__ __forceinline__ void wait_async0() {
  asm volatile("s_wait_asynccnt 0x0" ::: "memory");
}
// low 32 bits of a flat shared pointer == LDS byte offset (aperture in hi half)
__device__ __forceinline__ unsigned lds_off_of(const void* p) {
  return (unsigned)(unsigned long long)p;
}

// ---------------------------------------------------------------------------
// Prep 1: x (fp32, [ci=16][32][32][32][32]) -> channels-last bf16
//         xcl[d1][d2][d3][d4][ci]  (so every (i1,i2) slab is contiguous 32KB)
// ---------------------------------------------------------------------------
__global__ __launch_bounds__(256) void prep_x_kernel(
    const float* __restrict__ x, unsigned short* __restrict__ xcl) {
  const int p = blockIdx.x * 256 + threadIdx.x;     // (d1,d2,d3,d4) flat, < 2^20
  unsigned short e[16];
#pragma unroll
  for (int ci = 0; ci < 16; ++ci)
    e[ci] = f2bf(x[(size_t)ci * 1048576u + (unsigned)p]);
  v4u lo, hi;
#pragma unroll
  for (int q = 0; q < 4; ++q) {
    lo[q] = (unsigned)e[2*q]     | ((unsigned)e[2*q+1]     << 16);
    hi[q] = (unsigned)e[8 + 2*q] | ((unsigned)e[8 + 2*q+1] << 16);
  }
  v4u* dst = (v4u*)(xcl + (size_t)p * 16u);
  dst[0] = lo; dst[1] = hi;
}

// ---------------------------------------------------------------------------
// Prep 2/3: swizzle weights into the documented A-matrix (16x32 bf16) register
// layout, with k4 padded 3->4 (zero weights) so K per tap = 4*Ci (mult. of 32).
// Layout: A[mtile][kstep][lane][slot(=vgpr*2+half)], 16 bf16 per lane = 32B.
// Lane L: M = L%16, g = L/16; K(slot) = (v<4?0:16) + g*8 + 2*(v&3) + h.
// K within tap = k4*Ci + ci.
// ---------------------------------------------------------------------------
__device__ __forceinline__ int slotK(int lane, int slot) {
  const int v = slot >> 1, h = slot & 1, g = lane >> 4;
  return ((v < 4) ? 0 : 16) + g * 8 + 2 * (v & 3) + h;
}

__global__ __launch_bounds__(256) void prep_w1_kernel(
    const float* __restrict__ w1, unsigned short* __restrict__ A1) {
  const int idx = blockIdx.x * 256 + threadIdx.x;   // < 4*54*32*16 = 110592
  const int m    = idx / 27648;                     // M-tile (co/16)
  const int r0   = idx % 27648;
  const int t    = r0 / 512;                        // K-step, 0..53
  const int lane = (r0 >> 4) & 31;
  const int slot = r0 & 15;
  const int K    = slotK(lane, slot);
  const int tap  = t >> 1, s = t & 1;               // 2 K-steps per tap (Ci=16)
  const int k3 = tap % 3, k2 = (tap / 3) % 3, k1 = tap / 9;
  const int Kin = s * 32 + K;
  const int k4 = Kin >> 4, ci = Kin & 15;
  const int co = m * 16 + (lane & 15);
  float wv = 0.0f;
  if (k4 < 3) wv = w1[((((co*16 + ci)*3 + k1)*3 + k2)*3 + k3)*3 + k4];
  A1[idx] = f2bf(wv);
}

__global__ __launch_bounds__(256) void prep_w2_kernel(
    const float* __restrict__ w2, unsigned short* __restrict__ A2) {
  const int idx = blockIdx.x * 256 + threadIdx.x;   // < 216*32*16 = 110592
  const int t    = idx / 512;                       // K-step, 0..215
  const int lane = (idx >> 4) & 31;
  const int slot = idx & 15;
  const int K    = slotK(lane, slot);
  const int tap  = t >> 3, s = t & 7;               // 8 K-steps per tap (Ci=64)
  const int k3 = tap % 3, k2 = (tap / 3) % 3, k1 = tap / 9;
  const int Kin = s * 32 + K;
  const int k4 = Kin >> 6, ci = Kin & 63;
  const int co = lane & 15;
  float wv = 0.0f;
  if (k4 < 3) wv = w2[((((co*64 + ci)*3 + k1)*3 + k2)*3 + k3)*3 + k4];
  A2[idx] = f2bf(wv);
}

// ---------------------------------------------------------------------------
// Conv1: GEMM M=64(Co) N=65536(16^4 pos) K=1728(padded).
// One WG per (o1,o2). 8 waves: wave -> Mtile (w&3), Ntile group ((w>>2)*8..+7).
// Double-buffered LDS slab [i3p:34][i4p:36][ci:16] bf16 (2 x 39168 B),
// pad strips zeroed once; real region re-filled by async DMA each iteration.
// B frag per lane: 2x ds_load_b128 (ci contiguous). A frag: 2x global b128.
// Epilogue: bias + exact GELU -> bf16 channels-last h.
// ---------------------------------------------------------------------------
__global__ __launch_bounds__(256) void conv1_kernel(
    const unsigned short* __restrict__ xcl,
    const unsigned short* __restrict__ A1,
    const float* __restrict__ b1,
    unsigned short* __restrict__ hcl) {
  __shared__ v4u slab[2][2448];                     // 2 x 34*36*32B
  const int tid  = threadIdx.x;
  const int lane = tid & 31;
  const int wave = tid >> 5;
  const int o1 = blockIdx.x >> 4;
  const int o2 = blockIdx.x & 15;
  const int m     = wave & 3;
  const int jbase = (wave >> 2) * 8;
  const int n = lane & 15;
  const int g = lane >> 4;

  v8f acc[8];
#pragma unroll
  for (int j = 0; j < 8; ++j)
#pragma unroll
    for (int r = 0; r < 8; ++r) acc[j][r] = 0.0f;

  for (int i = tid; i < 2 * 2448; i += 256) {       // zero once: pads persist
    v4u z; z[0] = 0u; z[1] = 0u; z[2] = 0u; z[3] = 0u;
    slab[0][i] = z;                                 // flat over both buffers
  }
  __syncthreads();                                  // dscnt drained + barrier

  // uniform list of in-bounds (i1,i2) slabs
  int nv = 0;
  int lsrc[9], ltap[9];
  for (int k1 = 0; k1 < 3; ++k1) {
    const int i1 = 2*o1 - 1 + k1;
    if ((unsigned)i1 >= 32u) continue;
    for (int k2 = 0; k2 < 3; ++k2) {
      const int i2 = 2*o2 - 1 + k2;
      if ((unsigned)i2 >= 32u) continue;
      lsrc[nv] = i1*32 + i2;
      ltap[nv] = (k1*3 + k2)*3;                     // tap base (k3 added later)
      ++nv;
    }
  }

  // per-thread fill geometry (8 x b128 per thread = 32KB per slab)
  unsigned ldsoff[8], goff[8];
#pragma unroll
  for (int c = 0; c < 8; ++c) {
    const int idx  = tid + c * 256;
    const int d3   = idx >> 6;
    const int rem  = idx & 63;
    const int d4   = rem >> 1;
    const int half = rem & 1;
    ldsoff[c] = (unsigned)((((d3 + 1) * 36 + (d4 + 1)) * 2 + half) * 16);
    goff[c]   = (unsigned)(idx * 16);
  }
  const unsigned base0 = lds_off_of(&slab[0][0]);
  const unsigned base1 = lds_off_of(&slab[1][0]);

  // prologue: async-fill buffer 0 with first slab
  {
    const char* src = (const char*)(xcl + (size_t)lsrc[0] * 16384u);
#pragma unroll
    for (int c = 0; c < 8; ++c)
      async_b128(base0 + ldsoff[c], src + goff[c]);
  }

  for (int it = 0; it < nv; ++it) {
    const int cur = it & 1;
    wait_async0();                                  // own DMA landed in LDS
    __syncthreads();                                // everyone's DMA visible
    if (it + 1 < nv) {                              // prefetch next slab
      const unsigned nb = (cur ? base0 : base1);
      const char* src = (const char*)(xcl + (size_t)lsrc[it + 1] * 16384u);
#pragma unroll
      for (int c = 0; c < 8; ++c)
        async_b128(nb + ldsoff[c], src + goff[c]);
    }
    const v4u* __restrict__ sb = slab[cur];
    const int tapb = ltap[it];
    for (int k3 = 0; k3 < 3; ++k3) {
#pragma unroll
      for (int s = 0; s < 2; ++s) {
        const int t = (tapb + k3)*2 + s;
        BFrag a;
        const v4u* ap =
            (const v4u*)(A1 + (size_t)(((m*54 + t)*32 + lane) * 16));
        a.u[0] = ap[0]; a.u[1] = ap[1];
#pragma unroll
        for (int j = 0; j < 8; ++j) {
          const int i3p  = 2 * (jbase + j) + k3;
          const int bidx = (i3p * 36 + 2*n + 2*s) * 2 + g;
          BFrag b;
          b.u[0] = sb[bidx];                        // K 0-7 / 8-15 (k4=2s)
          b.u[1] = sb[bidx + 2];                    // K 16-23 / 24-31 (k4=2s+1)
          acc[j] = __builtin_amdgcn_wmma_f32_16x16x32_bf16(
                       false, a.v, false, b.v, (short)0, acc[j], false, false);
        }
      }
    }
  }

  // epilogue: bias + exact GELU, store 8 contiguous co as one b128
  const int cobase = m * 16 + g * 8;
  float bias[8];
#pragma unroll
  for (int r = 0; r < 8; ++r) bias[r] = b1[cobase + r];
#pragma unroll
  for (int j = 0; j < 8; ++j) {
    const int o3 = jbase + j;
    unsigned short e[8];
#pragma unroll
    for (int r = 0; r < 8; ++r) {
      const float xv = acc[j][r] + bias[r];
      const float ge = 0.5f * xv * (1.0f + erff(xv * 0.70710678118654752f));
      e[r] = f2bf(ge);
    }
    v4u pk;
#pragma unroll
    for (int q = 0; q < 4; ++q)
      pk[q] = (unsigned)e[2*q] | ((unsigned)e[2*q+1] << 16);
    const size_t off =
        ((size_t)(((o1*16 + o2)*16 + o3)*16 + n)) * 64u + (unsigned)cobase;
    *(v4u*)(hcl + off) = pk;
  }
}

// ---------------------------------------------------------------------------
// Conv2: GEMM M=16 N=4096(8^4) K=6912(padded). One WG per (o1,o2), 4 waves,
// one 16x16 output tile each. Double-buffered slab [i3p:18][i4p:18][ci:64]
// (2 x 41472 B). Output fp32 in reference layout out[co][o1][o2][o3][o4].
// ---------------------------------------------------------------------------
__global__ __launch_bounds__(128) void conv2_kernel(
    const unsigned short* __restrict__ hcl,
    const unsigned short* __restrict__ A2,
    const float* __restrict__ b2,
    float* __restrict__ out) {
  __shared__ v4u slab[2][2592];                     // 2 x 18*18*128B
  const int tid  = threadIdx.x;
  const int lane = tid & 31;
  const int wave = tid >> 5;                        // N-tile
  const int o1 = blockIdx.x >> 3;
  const int o2 = blockIdx.x & 7;
  const int n = lane & 15;
  const int g = lane >> 4;
  const int o3 = wave * 2 + (n >> 3);
  const int o4 = n & 7;

  v8f acc;
#pragma unroll
  for (int r = 0; r < 8; ++r) acc[r] = 0.0f;

  for (int i = tid; i < 2 * 2592; i += 128) {
    v4u z; z[0] = 0u; z[1] = 0u; z[2] = 0u; z[3] = 0u;
    slab[0][i] = z;
  }
  __syncthreads();

  int nv = 0;
  int lsrc[9], ltap[9];
  for (int k1 = 0; k1 < 3; ++k1) {
    const int i1 = 2*o1 - 1 + k1;
    if ((unsigned)i1 >= 16u) continue;
    for (int k2 = 0; k2 < 3; ++k2) {
      const int i2 = 2*o2 - 1 + k2;
      if ((unsigned)i2 >= 16u) continue;
      lsrc[nv] = i1*16 + i2;
      ltap[nv] = (k1*3 + k2)*3;
      ++nv;
    }
  }

  unsigned ldsoff[16], goff[16];
#pragma unroll
  for (int c = 0; c < 16; ++c) {
    const int idx  = tid + c * 128;
    const int d3   = idx >> 7;
    const int rem  = idx & 127;
    const int d4   = rem >> 3;
    const int part = rem & 7;
    ldsoff[c] = (unsigned)((((d3 + 1) * 18 + (d4 + 1)) * 8 + part) * 16);
    goff[c]   = (unsigned)(idx * 16);
  }
  const unsigned base0 = lds_off_of(&slab[0][0]);
  const unsigned base1 = lds_off_of(&slab[1][0]);

  {
    const char* src = (const char*)(hcl + (size_t)lsrc[0] * 16384u);
#pragma unroll
    for (int c = 0; c < 16; ++c)
      async_b128(base0 + ldsoff[c], src + goff[c]);
  }

  for (int it = 0; it < nv; ++it) {
    const int cur = it & 1;
    wait_async0();
    __syncthreads();
    if (it + 1 < nv) {
      const unsigned nb = (cur ? base0 : base1);
      const char* src = (const char*)(hcl + (size_t)lsrc[it + 1] * 16384u);
#pragma unroll
      for (int c = 0; c < 16; ++c)
        async_b128(nb + ldsoff[c], src + goff[c]);
    }
    const v4u* __restrict__ sb = slab[cur];
    const int tapb = ltap[it];
    for (int k3 = 0; k3 < 3; ++k3) {
      const int i3p = 2 * o3 + k3;
#pragma unroll
      for (int s = 0; s < 8; ++s) {
        const int t = (tapb + k3)*8 + s;
        BFrag a;
        const v4u* ap = (const v4u*)(A2 + (size_t)((t*32 + lane) * 16));
        a.u[0] = ap[0]; a.u[1] = ap[1];
        const int bidx =
            (i3p * 18 + 2*o4 + (s >> 1)) * 8 + (s & 1) * 4 + g;
        BFrag b;
        b.u[0] = sb[bidx];                          // ci chunk for K 0-15
        b.u[1] = sb[bidx + 2];                      // +16 ci for K 16-31
        acc = __builtin_amdgcn_wmma_f32_16x16x32_bf16(
                  false, a.v, false, b.v, (short)0, acc, false, false);
      }
    }
  }

#pragma unroll
  for (int r = 0; r < 8; ++r) {
    const int co = g * 8 + r;
    const float v = acc[r] + b2[co];
    out[(size_t)co * 4096u + (size_t)o1 * 512u + o2 * 64u + o3 * 8u + o4] = v;
  }
}

// ---------------------------------------------------------------------------
// Workspace layout (bytes, 256-aligned):
//   xcl : 0          .. 33,554,432   (16,777,216 bf16)
//   A1  : 33,554,432 .. +221,184
//   A2  : 33,775,616 .. +221,184
//   hcl : 33,996,800 .. +8,388,608   (total ~42.4 MB)
// ---------------------------------------------------------------------------
extern "C" void kernel_launch(void* const* d_in, const int* in_sizes, int n_in,
                              void* d_out, int out_size, void* d_ws, size_t ws_size,
                              hipStream_t stream) {
  const float* x  = (const float*)d_in[0];
  const float* w1 = (const float*)d_in[1];
  const float* b1 = (const float*)d_in[2];
  const float* w2 = (const float*)d_in[3];
  const float* b2 = (const float*)d_in[4];
  float* out = (float*)d_out;

  char* ws = (char*)d_ws;
  unsigned short* xcl = (unsigned short*)(ws);
  unsigned short* A1  = (unsigned short*)(ws + 33554432u);
  unsigned short* A2  = (unsigned short*)(ws + 33554432u + 221184u);
  unsigned short* hcl = (unsigned short*)(ws + 33554432u + 2u * 221184u);

  prep_x_kernel <<<4096, 256, 0, stream>>>(x, xcl);
  prep_w1_kernel<<< 432, 256, 0, stream>>>(w1, A1);
  prep_w2_kernel<<< 432, 256, 0, stream>>>(w2, A2);
  conv1_kernel  <<< 256, 256, 0, stream>>>(xcl, A1, b1, hcl);
  conv2_kernel  <<<  64, 128, 0, stream>>>(hcl, A2, b2, out);
}
